// LocalRGBPatchEmbedding_56006373540201
// MI455X (gfx1250) — matrix operand order repaired and checked by reading, not compiled, not run on hardware
//
#include <hip/hip_runtime.h>
#include <hip/hip_bf16.h>

typedef __attribute__((ext_vector_type(16))) _Float16 v16h;
typedef __attribute__((ext_vector_type(8)))  _Float16 v8h;
typedef __attribute__((ext_vector_type(8)))  float    v8f;

// Problem constants (from reference)
#define BB   16
#define TT   16
#define HH   384
#define WW   384
#define NN   2048
#define EMB  128
#define PP   9
#define HP   4
#define C1   64

// GEMM packing: K per conv1-channel padded 81 -> 96 so K is a multiple of 32
#define KPAD      96
#define KTOT      (C1 * KPAD)          // 6144
#define STEPS     (KTOT / 32)          // 192 WMMA K-steps total
#define CHUNK_CH  8                    // conv1 channels per pipeline chunk
#define CHUNK_K   (CHUNK_CH * KPAD)    // 768
#define CHUNK_ST  (CHUNK_K / 32)       // 24 WMMA steps per chunk
#define NCHUNK    (C1 / CHUNK_CH)      // 8

// Bpack layout: [gstep 0..191][ntile 0..7][lane 0..31][j 0..15] f16
//   n = ntile*16 + (lane&15)
//   K = gstep*32 + (lane>>4)*16 + j     (B-matrix: K split across lane halves)
#define BPACK_ELEMS (STEPS * 8 * 32 * 16)   // 786432 halves (1.5 MB)

__global__ __launch_bounds__(256)
void pack_w2_kernel(const float* __restrict__ W2, _Float16* __restrict__ Bpack)
{
    int idx = blockIdx.x * 256 + threadIdx.x;
    if (idx >= BPACK_ELEMS) return;
    int j     = idx & 15;
    int lane  = (idx >> 4) & 31;
    int ntile = (idx >> 9) & 7;
    int gstep = idx >> 12;
    int n = ntile * 16 + (lane & 15);
    int K = gstep * 32 + (lane >> 4) * 16 + j;
    int ci = K / KPAD;
    int r  = K - ci * KPAD;
    float val = 0.0f;
    if (r < 81) {
        int ky = r / 9, kx = r - ky * 9;
        val = W2[(((size_t)n * C1 + ci) * PP + ky) * PP + kx];
    }
    Bpack[idx] = (_Float16)val;
}

__global__ __launch_bounds__(128)
void patch_embed_kernel(const float* __restrict__ images,
                        const float* __restrict__ coords,
                        const int*   __restrict__ t_src,
                        const float* __restrict__ W1,
                        const float* __restrict__ b1,
                        const float* __restrict__ b2,
                        const _Float16* __restrict__ Bpack,
                        float* __restrict__ out)
{
    __shared__ float pat[16][3][11][11];                       // zero-padded patches (23.2 KB)
    __shared__ __attribute__((aligned(32))) _Float16 hbuf[16][CHUNK_K]; // conv1 out chunk (24.6 KB)
    __shared__ float w1s[C1 * 27];                             // 6.9 KB
    __shared__ float b1s[C1];
    __shared__ int   s_u[16], s_v[16], s_t[16];

    const int tid  = threadIdx.x;
    const int lane = tid & 31;
    const int wave = tid >> 5;
    const int tile = blockIdx.x;       // 2048 tiles of 16 patches

    // ---- Phase 0: preload weights, per-patch coords; zero padded patch LDS ----
    for (int i = tid; i < C1 * 27; i += 128) w1s[i] = W1[i];
    for (int i = tid; i < C1; i += 128)      b1s[i] = b1[i];
    if (tid < 16) {
        int p = tile * 16 + tid;
        int b = p >> 11;               // N == 2048
        int n = p & (NN - 1);
        float cx = coords[((size_t)(b * NN + n)) * 2 + 0];
        float cy = coords[((size_t)(b * NN + n)) * 2 + 1];
        int u = (int)rintf(cx * (float)WW); u = min(max(u, HP), WW - HP - 1);
        int v = (int)rintf(cy * (float)HH); v = min(max(v, HP), HH - HP - 1);
        int t = t_src[b * NN + n];          t = min(max(t, 0), TT - 1);
        s_u[tid] = u; s_v[tid] = v; s_t[tid] = t;
    }
    float* patf = &pat[0][0][0][0];
    for (int i = tid; i < 16 * 3 * 11 * 11; i += 128) patf[i] = 0.0f;
    __syncthreads();

    // ---- Phase 1: gather 16 patches (3x9x9 interior of padded 3x11x11) ----
    for (int idx = tid; idx < 16 * 3 * 81; idx += 128) {
        int m   = idx / 243;
        int rem = idx - m * 243;
        int c   = rem / 81;
        int s   = rem - c * 81;
        int y   = s / 9, x = s - y * 9;
        int p = tile * 16 + m;
        int b = p >> 11;
        size_t src = ((((size_t)b * TT + s_t[m]) * 3 + c) * HH + (s_v[m] - HP + y)) * WW
                     + (s_u[m] - HP + x);
        pat[m][c][y + 1][x + 1] = images[src];
    }

    // ---- Phase 2: chunked conv1 (VALU fp32) -> LDS f16 -> WMMA GEMM ----
    v8f acc0 = {};
    v8f acc1 = {};
    const int nt0 = wave;            // this wave's N-tiles (EMB channels nt*16..nt*16+15)
    const int nt1 = wave + 4;
    const int laneHalf = lane >> 4;
    const int mrow = lane & 15;

    for (int cg = 0; cg < NCHUNK; ++cg) {
        __syncthreads();   // pat ready (cg==0) / previous chunk's WMMA reads done

        // conv1 + ReLU for channels [cg*8, cg*8+8) -> hbuf[m][coL*96 + s] (s>=81 zero pad)
        for (int idx = tid; idx < 16 * CHUNK_K; idx += 128) {
            int m   = idx / CHUNK_K;
            int r   = idx - m * CHUNK_K;
            int coL = r / KPAD;
            int s   = r - coL * KPAD;
            _Float16 hv = (_Float16)0.0f;
            if (s < 81) {
                int y = s / 9, x = s - y * 9;
                int co = cg * CHUNK_CH + coL;
                float a = b1s[co];
                const float* wp = &w1s[co * 27];
                #pragma unroll
                for (int ci = 0; ci < 3; ++ci)
                    #pragma unroll
                    for (int ky = 0; ky < 3; ++ky)
                        #pragma unroll
                        for (int kx = 0; kx < 3; ++kx)
                            a = fmaf(pat[m][ci][y + ky][x + kx], wp[ci * 9 + ky * 3 + kx], a);
                a = fmaxf(a, 0.0f);
                hv = (_Float16)a;
            }
            hbuf[m][r] = hv;
        }
        __syncthreads();   // hbuf ready

        // WMMA: 24 K-steps of 16x16x32, each wave does 2 N-tiles
        const _Float16* bbase = Bpack + (size_t)(cg * CHUNK_ST) * (8 * 32 * 16);
        #pragma unroll 2
        for (int st = 0; st < CHUNK_ST; ++st) {
            // A-matrix in contiguous k order (same order as Bpack):
            //   lanes 0-15 : M=lane,  VGPR0-7 = K[st*32+0..15]
            //   lanes 16-31: M=l-16,  VGPR0-7 = K[st*32+16..31]
            int k0 = st * 32 + laneHalf * 16;
            const _Float16* ap = &hbuf[mrow][k0];
            v8h alo = *(const v8h*)ap;          // ds_load_b128
            v8h ahi = *(const v8h*)(ap + 8);    // ds_load_b128
            v16h A = __builtin_shufflevector(alo, ahi,
                                             0, 1, 2, 3, 4, 5, 6, 7,
                                             8, 9, 10, 11, 12, 13, 14, 15);

            const _Float16* bp0 = bbase + ((size_t)st * 8 + nt0) * 512 + lane * 16;
            const _Float16* bp1 = bbase + ((size_t)st * 8 + nt1) * 512 + lane * 16;
            v16h B0 = *(const v16h*)bp0;        // 32B/lane -> global_load_b128 x2
            v16h B1 = *(const v16h*)bp1;

            acc0 = __builtin_amdgcn_wmma_f32_16x16x32_f16(
                       false, A, false, B0, (short)0, acc0, false, false);
            acc1 = __builtin_amdgcn_wmma_f32_16x16x32_f16(
                       false, A, false, B1, (short)0, acc1, false, false);
        }
    }

    // ---- Phase 3: bias + store (C/D layout: lane 0-15 -> M=r, lane 16-31 -> M=r+8) ----
    int col = lane & 15;
    int ch0 = nt0 * 16 + col;
    int ch1 = nt1 * 16 + col;
    float bb0 = b2[ch0];
    float bb1 = b2[ch1];
    #pragma unroll
    for (int r = 0; r < 8; ++r) {
        int M = r + 8 * laneHalf;
        size_t p = (size_t)tile * 16 + M;
        out[p * EMB + ch0] = acc0[r] + bb0;
        out[p * EMB + ch1] = acc1[r] + bb1;
    }
}

extern "C" void kernel_launch(void* const* d_in, const int* in_sizes, int n_in,
                              void* d_out, int out_size, void* d_ws, size_t ws_size,
                              hipStream_t stream)
{
    const float* images = (const float*)d_in[0];
    const float* coords = (const float*)d_in[1];
    const int*   t_src  = (const int*)d_in[2];
    const float* W1     = (const float*)d_in[3];
    const float* b1     = (const float*)d_in[4];
    const float* W2     = (const float*)d_in[5];
    const float* b2     = (const float*)d_in[6];
    float* out = (float*)d_out;
    _Float16* Bpack = (_Float16*)d_ws;   // needs BPACK_ELEMS * 2 = 1.5 MB of scratch

    pack_w2_kernel<<<(BPACK_ELEMS + 255) / 256, 256, 0, stream>>>(W2, Bpack);

    int n_tiles = (BB * NN) / 16;        // 2048 workgroups of 16 patches
    patch_embed_kernel<<<n_tiles, 128, 0, stream>>>(images, coords, t_src,
                                                    W1, b1, b2, Bpack, out);
}
